// ResidualVectorQuantizer_64278480552689
// MI455X (gfx1250) — compile-verified, matrix operand order checked by reading, not run on hardware
//
#include <hip/hip_runtime.h>
#include <hip/hip_bf16.h>

// Problem constants (from reference): B=16, Q=8, W=256, D=256, K=1024, NQ=8
#define RVQ_N   32768      // B*Q*W
#define RVQ_D   256
#define RVQ_K   1024
#define RVQ_NQ  8

typedef __attribute__((ext_vector_type(16))) __bf16 v16bf;
typedef __attribute__((ext_vector_type(8)))  __bf16 v8bf;
typedef __attribute__((ext_vector_type(8)))  float  v8f;

// ---------------------------------------------------------------------------
// Prep kernels
// ---------------------------------------------------------------------------

// residual = z (fp32 working copy), qtot (d_out quantized region) = 0
__global__ void rvq_init_kernel(const float* __restrict__ z,
                                float* __restrict__ res,
                                float* __restrict__ qtot,
                                float* __restrict__ closs_acc) {
    size_t i = (size_t)blockIdx.x * blockDim.x + threadIdx.x;
    if (i < (size_t)RVQ_N * RVQ_D) {
        res[i]  = z[i];
        qtot[i] = 0.0f;
    }
    if (i == 0) *closs_acc = 0.0f;
}

// bf16 copy of all codebooks (done once per call, reused by all stages)
__global__ void rvq_cvt_cb_kernel(const float* __restrict__ cb,
                                  __bf16* __restrict__ cbb) {
    size_t i = (size_t)blockIdx.x * blockDim.x + threadIdx.x;
    if (i < (size_t)RVQ_NQ * RVQ_K * RVQ_D) cbb[i] = (__bf16)cb[i];
}

// cnorm[q][k] = sum_d cb[q][k][d]^2   (one block per (q,k), 256 threads)
__global__ void rvq_cnorm_kernel(const float* __restrict__ cb,
                                 float* __restrict__ cnorm) {
    __shared__ float sm[RVQ_D];
    int qk = blockIdx.x;                 // 0 .. NQ*K-1
    int t  = threadIdx.x;                // 0 .. D-1
    float v = cb[(size_t)qk * RVQ_D + t];
    sm[t] = v * v;
    __syncthreads();
    for (int s = RVQ_D / 2; s > 0; s >>= 1) {
        if (t < s) sm[t] += sm[t + s];
        __syncthreads();
    }
    if (t == 0) cnorm[qk] = sm[0];
}

// ---------------------------------------------------------------------------
// Stage GEMM + argmin:  score[n][k] = cnorm[k] - 2 * r_n . c_k   (WMMA bf16)
// Each wave owns TWO 16-row strips (32 rows) and reuses every B fragment for
// two WMMAs -> 0.5 KB of B traffic per WMMA instead of 1 KB.
// 4 waves / block, 256 blocks -> 32768 rows.  K swept in 64 tiles of 16.
// ---------------------------------------------------------------------------

#define WAVES_PER_BLOCK 4
#define STRIPS_PER_WAVE 2

// Load one 16x32 bf16 A fragment (rows rbase..rbase+15, K-dim d = kk*32..+31)
// from the fp32 residual, converting in registers.  Lane layout (ISA 7.12.2):
// lanes 0-15 hold row M=lane, elems {d0..d0+7, d0+16..d0+23};
// lanes 16-31 hold row M=lane-16, elems {d0+8..d0+15, d0+24..d0+31}.
__device__ __forceinline__ v16bf load_a_frag(const float* __restrict__ rptr,
                                             int kk, int hsel) {
    const int d0 = kk * 32 + hsel * 8;
    const float4* p0 = (const float4*)(rptr + d0);
    const float4* p1 = (const float4*)(rptr + d0 + 16);
    float4 a0 = p0[0], a1 = p0[1];
    float4 b0 = p1[0], b1 = p1[1];
    v16bf a;
    a[0]  = (__bf16)a0.x; a[1]  = (__bf16)a0.y; a[2]  = (__bf16)a0.z; a[3]  = (__bf16)a0.w;
    a[4]  = (__bf16)a1.x; a[5]  = (__bf16)a1.y; a[6]  = (__bf16)a1.z; a[7]  = (__bf16)a1.w;
    a[8]  = (__bf16)b0.x; a[9]  = (__bf16)b0.y; a[10] = (__bf16)b0.z; a[11] = (__bf16)b0.w;
    a[12] = (__bf16)b1.x; a[13] = (__bf16)b1.y; a[14] = (__bf16)b1.z; a[15] = (__bf16)b1.w;
    return a;
}

__global__ __launch_bounds__(32 * WAVES_PER_BLOCK)
void rvq_argmin_kernel(const float*  __restrict__ res,    // [N, D] fp32 residual
                       const __bf16* __restrict__ cbb,    // [K, D] bf16 codebook (stage q)
                       const float*  __restrict__ cnorm,  // [K]
                       int*          __restrict__ tok) {  // [N]
    const int lane    = threadIdx.x & 31;
    const int wave    = threadIdx.x >> 5;
    const int rowBase = (blockIdx.x * WAVES_PER_BLOCK + wave) * (16 * STRIPS_PER_WAVE);
    const int lanelo  = lane & 15;
    const int hsel    = lane >> 4;       // which K-half this lane holds

    // ---- Load the two 16x256 A-strips once, fp32 -> bf16 fragments ----
    v16bf A0[8], A1[8];
    {
        const float* r0 = res + (size_t)(rowBase + lanelo) * RVQ_D;
        const float* r1 = res + (size_t)(rowBase + 16 + lanelo) * RVQ_D;
        #pragma unroll
        for (int kk = 0; kk < 8; ++kk) A0[kk] = load_a_frag(r0, kk, hsel);
        #pragma unroll
        for (int kk = 0; kk < 8; ++kk) A1[kk] = load_a_frag(r1, kk, hsel);
    }

    float minval0[8], minval1[8];
    int   minidx0[8], minidx1[8];
    #pragma unroll
    for (int j = 0; j < 8; ++j) {
        minval0[j] = 3.402823466e38f; minidx0[j] = 0;
        minval1[j] = 3.402823466e38f; minidx1[j] = 0;
    }

    // ---- Sweep 64 column tiles of 16 codewords each ----
    for (int ct = 0; ct < RVQ_K / 16; ++ct) {
        const int col = ct * 16 + lanelo;
        // B fragment: lane layout mirrors A with column = codebook row `col`
        const __bf16* bbase = cbb + (size_t)col * RVQ_D;
        v8f acc0 = {};
        v8f acc1 = {};
        #pragma unroll
        for (int kk = 0; kk < 8; ++kk) {
            const int d0 = kk * 32 + hsel * 8;
            v8bf blo = *(const v8bf*)(bbase + d0);
            v8bf bhi = *(const v8bf*)(bbase + d0 + 16);
            v16bf bf = __builtin_shufflevector(blo, bhi,
                        0, 1, 2, 3, 4, 5, 6, 7, 8, 9, 10, 11, 12, 13, 14, 15);
            // two WMMAs share one B fragment (independent accumulators)
            acc0 = __builtin_amdgcn_wmma_f32_16x16x32_bf16(
                       false, A0[kk], false, bf, (short)0, acc0, false, false);
            acc1 = __builtin_amdgcn_wmma_f32_16x16x32_bf16(
                       false, A1[kk], false, bf, (short)0, acc1, false, false);
        }
        const float cn = cnorm[col];
        #pragma unroll
        for (int j = 0; j < 8; ++j) {
            float s0 = __builtin_fmaf(-2.0f, acc0[j], cn);
            float s1 = __builtin_fmaf(-2.0f, acc1[j], cn);
            if (s0 < minval0[j]) { minval0[j] = s0; minidx0[j] = col; }
            if (s1 < minval1[j]) { minval1[j] = s1; minidx1[j] = col; }
        }
    }

    // ---- Reduce argmin across the 16 lanes that share a row ----
    #pragma unroll
    for (int off = 8; off >= 1; off >>= 1) {
        #pragma unroll
        for (int j = 0; j < 8; ++j) {
            float ov0 = __shfl_xor(minval0[j], off, 32);
            int   oi0 = __shfl_xor(minidx0[j], off, 32);
            if (ov0 < minval0[j] || (ov0 == minval0[j] && oi0 < minidx0[j])) {
                minval0[j] = ov0; minidx0[j] = oi0;
            }
            float ov1 = __shfl_xor(minval1[j], off, 32);
            int   oi1 = __shfl_xor(minidx1[j], off, 32);
            if (ov1 < minval1[j] || (ov1 == minval1[j] && oi1 < minidx1[j])) {
                minval1[j] = ov1; minidx1[j] = oi1;
            }
        }
    }
    // C-layout: lanes 0..15 hold rows M=j (VGPR j), lanes 16..31 rows M=j+8
    if (lane == 0) {
        #pragma unroll
        for (int j = 0; j < 8; ++j) {
            tok[rowBase + j]      = minidx0[j];
            tok[rowBase + 16 + j] = minidx1[j];
        }
    }
    if (lane == 16) {
        #pragma unroll
        for (int j = 0; j < 8; ++j) {
            tok[rowBase + 8 + j]  = minidx0[j];
            tok[rowBase + 24 + j] = minidx1[j];
        }
    }
}

// ---------------------------------------------------------------------------
// Stage update: r' = r - c[idx]; qtot += c[idx]; loss += sum(r'^2); token out
// One block per row, 256 threads = D
// ---------------------------------------------------------------------------
__global__ void rvq_update_kernel(float* __restrict__ res,        // [N, D]
                                  const float* __restrict__ cbq,  // [K, D] fp32 (stage q)
                                  const int* __restrict__ tok,    // [N]
                                  float* __restrict__ qtot,       // [N, D] (d_out)
                                  float* __restrict__ tok_out,    // [N, NQ] (d_out, as float)
                                  float* __restrict__ closs_acc,  // scalar
                                  int q) {
    __shared__ float sm[RVQ_D];
    const int n = blockIdx.x;
    const int t = threadIdx.x;
    const int idx = tok[n];
    const float c  = cbq[(size_t)idx * RVQ_D + t];
    const float r  = res[(size_t)n * RVQ_D + t];
    const float nr = r - c;
    res[(size_t)n * RVQ_D + t]   = nr;
    qtot[(size_t)n * RVQ_D + t] += c;
    sm[t] = nr * nr;
    __syncthreads();
    for (int s = RVQ_D / 2; s > 0; s >>= 1) {
        if (t < s) sm[t] += sm[t + s];
        __syncthreads();
    }
    if (t == 0) {
        atomicAdd(closs_acc, sm[0]);
        tok_out[(size_t)n * RVQ_NQ + q] = (float)idx;
    }
}

// closs = 1.25 * acc / (N*D*NQ)
__global__ void rvq_finish_kernel(const float* __restrict__ closs_acc,
                                  float* __restrict__ closs_out) {
    if (threadIdx.x == 0 && blockIdx.x == 0) {
        *closs_out = (*closs_acc) * (1.25f / ((float)RVQ_N * (float)RVQ_D * (float)RVQ_NQ));
    }
}

// ---------------------------------------------------------------------------
// Host-side launcher
// ---------------------------------------------------------------------------
extern "C" void kernel_launch(void* const* d_in, const int* in_sizes, int n_in,
                              void* d_out, int out_size, void* d_ws, size_t ws_size,
                              hipStream_t stream) {
    const float* z  = (const float*)d_in[0];   // [B,Q,W,D] = [N,D]
    const float* cb = (const float*)d_in[1];   // [NQ,K,D]

    // d_out layout: quantized [N*D] | tokens [N*NQ] (float) | closs [1]
    float* out_q    = (float*)d_out;
    float* out_tok  = out_q + (size_t)RVQ_N * RVQ_D;
    float* out_loss = out_tok + (size_t)RVQ_N * RVQ_NQ;

    // workspace layout
    char* ws = (char*)d_ws;
    float*  res      = (float*)ws;                                   // 32 MB
    ws += (size_t)RVQ_N * RVQ_D * sizeof(float);
    __bf16* cbb      = (__bf16*)ws;                                  // 4 MB
    ws += (size_t)RVQ_NQ * RVQ_K * RVQ_D * sizeof(__bf16);
    float*  cnorm    = (float*)ws;                                   // 32 KB
    ws += (size_t)RVQ_NQ * RVQ_K * sizeof(float);
    int*    tok      = (int*)ws;                                     // 128 KB
    ws += (size_t)RVQ_N * sizeof(int);
    float*  closs    = (float*)ws;                                   // 4 B

    // ---- prep ----
    {
        int total = RVQ_N * RVQ_D;
        rvq_init_kernel<<<(total + 255) / 256, 256, 0, stream>>>(z, res, out_q, closs);
    }
    {
        int total = RVQ_NQ * RVQ_K * RVQ_D;
        rvq_cvt_cb_kernel<<<(total + 255) / 256, 256, 0, stream>>>(cb, cbb);
    }
    rvq_cnorm_kernel<<<RVQ_NQ * RVQ_K, RVQ_D, 0, stream>>>(cb, cnorm);

    // ---- sequential stages ----
    const int rowsPerBlock = 16 * STRIPS_PER_WAVE * WAVES_PER_BLOCK;   // 128
    const int gemmBlocks   = RVQ_N / rowsPerBlock;                     // 256
    for (int q = 0; q < RVQ_NQ; ++q) {
        const __bf16* cbb_q = cbb + (size_t)q * RVQ_K * RVQ_D;
        const float*  cb_q  = cb  + (size_t)q * RVQ_K * RVQ_D;
        const float*  cn_q  = cnorm + (size_t)q * RVQ_K;

        rvq_argmin_kernel<<<gemmBlocks, 32 * WAVES_PER_BLOCK, 0, stream>>>(
            res, cbb_q, cn_q, tok);

        rvq_update_kernel<<<RVQ_N, RVQ_D, 0, stream>>>(
            res, cb_q, tok, out_q, out_tok, closs, q);
    }

    rvq_finish_kernel<<<1, 32, 0, stream>>>(closs, out_loss);
}
